// SS2D_31516470018291
// MI455X (gfx1250) — matrix operand order repaired and checked by reading, not compiled, not run on hardware
//
#include <hip/hip_runtime.h>
#include <hip/hip_bf16.h>
#include <math.h>
#include <stdint.h>

// ---------------- problem constants ----------------
#define SS_B    4
#define SS_K    4
#define SS_D    192      // d_inner (== C)
#define SS_N    16       // d_state
#define SS_R    6        // dt_rank
#define SS_L    9216     // H*W
#define SS_F    152      // K * (R + 2N) = 4*38
#define SS_FT   10       // ceil(152/16) N-tiles
#define SS_MT   576      // L/16 M-tiles per batch
#define SS_S    36       // scan chunks (L = 36*256)
#define SS_CL   256      // chunk length
#define SS_T    32       // LDS tile (timesteps)
#define SS_UP   36       // padded u row stride (floats): 144B, 16B-aligned, 2-way bank

// workspace layout (floats)
#define WS_XDBL 0                      // (B,K,L,38)          5,603,328
#define WS_HEND 5603328                // (B,K,S,D,16)        1,769,472
#define WS_DSUM 7372800                // (B,K,S,D)             110,592
#define WS_H0   7483392                // (B,K,S,D,16)        1,769,472

typedef __attribute__((ext_vector_type(16))) _Float16 v16h;
typedef __attribute__((ext_vector_type(8)))  float    v8f;

__device__ __forceinline__ unsigned lds_off(const void* p) {
    // low 32 bits of a flat pointer into the LDS aperture == LDS byte offset
    return (unsigned)(uintptr_t)p;
}

__device__ __forceinline__ void async_b128(unsigned lds_addr, const void* gaddr) {
    asm volatile("global_load_async_to_lds_b128 %0, %1, off"
                 :: "v"(lds_addr), "v"((uint64_t)(uintptr_t)gaddr) : "memory");
}

__device__ __forceinline__ void wait_async0() {
    asm volatile("s_wait_asynccnt 0x0" ::: "memory");
}

// =====================================================================
// Kernel 1: x_dbl[b,k,l,j] = sum_c xl[b,l,c] * Wxp[k,j,c]
// One wave per 16(M) x 160(N) output strip: A tile (16x32 f16) loaded
// once per K-step and reused across all 10 N-tiles (10 accumulators).
// All global loads are base-pointer + compile-time IOFFSET:
//   A: lane base = x + (b*D + (hi?8:0))*L + l0+m ; offset (c0+kkLo)*L*4
//   B: per-nt row ptr absorbs (hi?8:0)           ; offset (c0+kkLo)*4
// Tail lanes of nt=9 clamp their weight row in-bounds; their D columns
// are never stored.
// =====================================================================
__global__ void __launch_bounds__(128)
proj_wmma(const float* __restrict__ x,      // (B, D, L)
          const float* __restrict__ Wxp,    // (K, 38, 192)
          float* __restrict__ xdbl)         // (B, K, L, 38)
{
    const int lane = threadIdx.x & 31;
    const int mt   = blockIdx.x * 4 + (threadIdx.x >> 5);  // M-tile 0..2303
    const int b    = mt / SS_MT;
    const int l0   = (mt % SS_MT) * 16;
    const int m    = lane & 15;
    const bool hi  = lane >= 16;

    const float* abase = x + ((size_t)b * SS_D + (hi ? 8 : 0)) * SS_L + (l0 + m);

    // per-N-tile weight row pointers (clamped in-bounds)
    const float* wrow[SS_FT];
    #pragma unroll
    for (int nt = 0; nt < SS_FT; ++nt) {
        const int f  = nt * 16 + (lane & 15);
        const int fc = (f < SS_F) ? f : (SS_F - 1);
        const int kq = fc / 38;
        const int j  = fc - 38 * kq;
        wrow[nt] = Wxp + (size_t)(kq * 38 + j) * SS_D + (hi ? 8 : 0);
    }

    v8f acc[SS_FT] = {};
    #pragma unroll
    for (int c0 = 0; c0 < SS_D; c0 += 32) {
        v16h a;
        #pragma unroll
        for (int i = 0; i < 16; ++i) {
            const int kkLo = (i < 8) ? i : (i + 8);
            a[i] = (_Float16)abase[(size_t)(c0 + kkLo) * SS_L];
        }
        #pragma unroll
        for (int nt = 0; nt < SS_FT; ++nt) {
            v16h bm;
            #pragma unroll
            for (int i = 0; i < 16; ++i)
                bm[i] = (_Float16)wrow[nt][c0 + ((i < 8) ? i : (i + 8))];
            acc[nt] = __builtin_amdgcn_wmma_f32_16x16x32_f16(
                          false, a, false, bm, (short)0, acc[nt], false, false);
        }
    }

    #pragma unroll
    for (int nt = 0; nt < SS_FT; ++nt) {
        const int f = nt * 16 + (lane & 15);
        if (f < SS_F) {                       // uniform-true except nt==9
            const int kq = f / 38;
            const int j  = f - 38 * kq;
            float* dst = xdbl + (((size_t)(b * SS_K + kq)) * SS_L + l0) * 38 + j;
            #pragma unroll
            for (int v = 0; v < 8; ++v)
                dst[(size_t)(v + (hi ? 8 : 0)) * 38] = acc[nt][v];
        }
    }
}

// =====================================================================
// Chunked selective scan.
//  pass1 (PASS3=false): scan chunk from h=0, emit (h_end, sum(delta))
//  pass3 (PASS3=true) : scan chunk from stored h0, emit y
// block = 192 threads (one per d); grid = B*K*S
// Staging uses CDNA5 async copy: global_load_async_to_lds_b128,
// completion via s_wait_asynccnt then workgroup barrier.
// =====================================================================
template <bool PASS3>
__global__ void __launch_bounds__(192)
scan_kernel(const float* __restrict__ x,     // (B,D,L)  -> u rows
            const float* __restrict__ xdbl,  // (B,K,L,38)
            const float* __restrict__ Wdt,   // (K,D,6)
            const float* __restrict__ bias,  // (K,D)
            const float* __restrict__ Alogs, // (K*D,16)
            const float* __restrict__ Ds,    // (K*D)
            float* __restrict__ hend,        // (B,K,S,D,16)
            float* __restrict__ dsum,        // (B,K,S,D)
            const float* __restrict__ h0,    // (B,K,S,D,16)
            float* __restrict__ out)         // (K,B,D,L)
{
    __shared__ __attribute__((aligned(16))) float lds_u [SS_D * SS_UP]; // [d][t] padded
    __shared__ __attribute__((aligned(16))) float lds_xd[SS_T * 38];
    __shared__ __attribute__((aligned(16))) float lds_y [PASS3 ? (SS_T * (SS_D + 1)) : 1];

    const int tid  = threadIdx.x;
    const int d    = tid;
    const int cidx = blockIdx.x % SS_S;
    const int k    = (blockIdx.x / SS_S) % SS_K;
    const int b    = blockIdx.x / (SS_S * SS_K);
    const int bk   = b * SS_K + k;

    // per-(k,d) parameters
    float wdt[SS_R];
    #pragma unroll
    for (int r = 0; r < SS_R; ++r) wdt[r] = Wdt[(k * SS_D + d) * SS_R + r];
    const float bi  = bias[k * SS_D + d];
    const float dsk = Ds[k * SS_D + d];
    float An[SS_N];
    #pragma unroll
    for (int nn = 0; nn < SS_N; ++nn)
        An[nn] = -__expf(Alogs[(k * SS_D + d) * SS_N + nn]);

    float h[SS_N];
    if (PASS3) {
        const float* hp = h0 + (((size_t)bk * SS_S + cidx) * SS_D + d) * SS_N;
        #pragma unroll
        for (int nn = 0; nn < SS_N; ++nn) h[nn] = hp[nn];
    } else {
        #pragma unroll
        for (int nn = 0; nn < SS_N; ++nn) h[nn] = 0.0f;
    }
    float deltasum = 0.0f;

    const int    t0c   = cidx * SS_CL;
    const float* ubase = x    + ((size_t)b * SS_D + d) * SS_L;
    const float* xdb   = xdbl + ((size_t)bk * SS_L) * 38;

    for (int tb = 0; tb < SS_CL; tb += SS_T) {
        const int tstart = t0c + tb;
        __syncthreads();   // prior tile's LDS reads complete before overwrite

        // async-stage u tile: thread d owns padded row d (8 x b128)
        #pragma unroll
        for (int c4 = 0; c4 < SS_T; c4 += 4)
            async_b128(lds_off(&lds_u[d * SS_UP + c4]), ubase + tstart + c4);
        // async-stage x_dbl tile: 304 contiguous 16B chunks
        for (int i = tid; i < (SS_T * 38) / 4; i += SS_D)
            async_b128(lds_off(&lds_xd[i * 4]), xdb + (size_t)tstart * 38 + i * 4);
        // prefetch next u tile into cache (global_prefetch_b8)
        if (tstart + SS_T < SS_L)
            __builtin_prefetch(ubase + tstart + SS_T, 0, 1);

        wait_async0();     // our async writes landed in LDS
        __syncthreads();   // everyone's writes visible

        for (int tt = 0; tt < SS_T; ++tt) {
            const float* row = lds_xd + tt * 38;   // [dts(6) | B(16) | C(16)]
            float z = bi;
            #pragma unroll
            for (int r = 0; r < SS_R; ++r) z = fmaf(row[r], wdt[r], z);
            const float delta = (z > 20.0f) ? z : log1pf(__expf(z));
            deltasum += delta;
            const float u  = lds_u[d * SS_UP + tt];
            const float du = delta * u;
            float y = 0.0f;
            #pragma unroll
            for (int nn = 0; nn < SS_N; ++nn) {
                const float dA = __expf(delta * An[nn]);
                h[nn] = fmaf(dA, h[nn], du * row[SS_R + nn]);
                if (PASS3) y = fmaf(h[nn], row[SS_R + SS_N + nn], y);
            }
            if (PASS3)
                lds_y[tt * (SS_D + 1) + d] = fmaf(dsk, u, y);
        }

        if (PASS3) {
            __syncthreads();
            // transposed coalesced store: out[((k*B+b)*D + c)*L + t]
            float* obase = out + ((size_t)(k * SS_B + b) * SS_D) * SS_L + tstart;
            for (int i = tid; i < SS_D * SS_T; i += SS_D) {
                const int c  = i / SS_T;
                const int xx = i % SS_T;
                obase[(size_t)c * SS_L + xx] = lds_y[xx * (SS_D + 1) + c];
            }
        }
    }

    if (!PASS3) {
        float* hp = hend + (((size_t)bk * SS_S + cidx) * SS_D + d) * SS_N;
        #pragma unroll
        for (int nn = 0; nn < SS_N; ++nn) hp[nn] = h[nn];
        dsum[((size_t)bk * SS_S + cidx) * SS_D + d] = deltasum;
    }
}

// =====================================================================
// Sequential combine across chunks per (b,k,d) lane.
// h0[c] = exp(An * dsum[c-1]) * h0[c-1] + hend[c-1]   (h0[0] = 0)
// =====================================================================
__global__ void __launch_bounds__(192)
scan_combine(const float* __restrict__ Alogs,
             const float* __restrict__ hend,
             const float* __restrict__ dsum,
             float* __restrict__ h0)
{
    const int g  = blockIdx.x * 192 + threadIdx.x;   // 0 .. B*K*D-1
    const int d  = g % SS_D;
    const int bk = g / SS_D;
    const int k  = bk % SS_K;

    float An[SS_N];
    #pragma unroll
    for (int nn = 0; nn < SS_N; ++nn)
        An[nn] = -__expf(Alogs[(k * SS_D + d) * SS_N + nn]);

    float h[SS_N];
    #pragma unroll
    for (int nn = 0; nn < SS_N; ++nn) h[nn] = 0.0f;

    for (int c = 0; c < SS_S; ++c) {
        const size_t base = ((size_t)bk * SS_S + c) * SS_D + d;
        float* hp = h0 + base * SS_N;
        #pragma unroll
        for (int nn = 0; nn < SS_N; ++nn) hp[nn] = h[nn];
        const float ds_ = dsum[base];
        const float* he = hend + base * SS_N;
        #pragma unroll
        for (int nn = 0; nn < SS_N; ++nn)
            h[nn] = fmaf(__expf(An[nn] * ds_), h[nn], he[nn]);
    }
}

// =====================================================================
extern "C" void kernel_launch(void* const* d_in, const int* in_sizes, int n_in,
                              void* d_out, int out_size, void* d_ws, size_t ws_size,
                              hipStream_t stream)
{
    const float* x     = (const float*)d_in[0];  // (B, D, H, W)
    const float* Wxp   = (const float*)d_in[1];  // (K, 38, D)
    const float* Wdt   = (const float*)d_in[2];  // (K, D, 6)
    const float* bias  = (const float*)d_in[3];  // (K, D)
    const float* Alogs = (const float*)d_in[4];  // (K*D, 16)
    const float* Ds    = (const float*)d_in[5];  // (K*D)
    float* out = (float*)d_out;                  // 4 x (B, D, H, W) concatenated
    float* ws  = (float*)d_ws;

    float* xdbl = ws + WS_XDBL;
    float* hend = ws + WS_HEND;
    float* dsum = ws + WS_DSUM;
    float* h0   = ws + WS_H0;

    // 1) projection GEMM via WMMA: 2304 M-strips, 4 waves/block
    proj_wmma<<<(SS_B * SS_MT) / 4, 128, 0, stream>>>(x, Wxp, xdbl);

    // 2) pass 1: per-chunk scan from zero state -> (h_end, sum delta)
    scan_kernel<false><<<SS_B * SS_K * SS_S, SS_D, 0, stream>>>(
        x, xdbl, Wdt, bias, Alogs, Ds, hend, dsum, nullptr, nullptr);

    // 3) sequential chunk combine -> per-chunk initial states
    scan_combine<<<(SS_B * SS_K * SS_D) / 192, 192, 0, stream>>>(
        Alogs, hend, dsum, h0);

    // 4) pass 3: re-scan each chunk from its true initial state, emit y
    scan_kernel<true><<<SS_B * SS_K * SS_S, SS_D, 0, stream>>>(
        x, xdbl, Wdt, bias, Alogs, Ds, nullptr, nullptr, h0, out);
}
